// GraphTransformerLayer_154618822904
// MI455X (gfx1250) — matrix operand order repaired and checked by reading, not compiled, not run on hardware
//
#include <hip/hip_runtime.h>

// ---------------- problem constants ----------------
#define NN   100000
#define EE   1600000
#define HH   128
#define NHH  8
#define HDD  16
#define FF   512          // 4*H

typedef __attribute__((ext_vector_type(16))) __bf16       v16bf;
typedef __attribute__((ext_vector_type(8)))  float        v8f;
typedef __attribute__((ext_vector_type(8)))  unsigned int v8u;

// ---------------- helpers ----------------
static __device__ __forceinline__ unsigned short f2bf(float f) {
  unsigned u = __builtin_bit_cast(unsigned, f);
  u += 0x7fffu + ((u >> 16) & 1u);          // round-to-nearest-even
  return (unsigned short)(u >> 16);
}

// monotonic float <-> uint mapping for atomicMax-based segment max
static __device__ __forceinline__ unsigned fenc(float f) {
  unsigned u = __builtin_bit_cast(unsigned, f);
  return (u & 0x80000000u) ? ~u : (u | 0x80000000u);
}
static __device__ __forceinline__ float fdec(unsigned u) {
  u = (u & 0x80000000u) ? (u & 0x7fffffffu) : ~u;
  return __builtin_bit_cast(float, u);
}

static __device__ __forceinline__ float wave_sum32(float s) {
#pragma unroll
  for (int off = 16; off > 0; off >>= 1) s += __shfl_xor(s, off, 32);
  return s;
}

// ---------------- conversions ----------------
__global__ void f32_to_bf16_kernel(const float* __restrict__ in,
                                   unsigned short* __restrict__ out, int n) {
  int i = blockIdx.x * blockDim.x + threadIdx.x;
  if (i < n) out[i] = f2bf(in[i]);
}

// W [K, Nout] row-major f32 -> Wt [Nout, K] bf16 (so B k-pairs are contiguous)
__global__ void transpose_to_bf16_kernel(const float* __restrict__ W,
                                         unsigned short* __restrict__ Wt,
                                         int K, int Nout) {
  int i = blockIdx.x * blockDim.x + threadIdx.x;
  if (i >= K * Nout) return;
  int k = i / Nout, n = i % Nout;
  Wt[(size_t)n * K + k] = f2bf(W[i]);
}

// ---------------- WMMA GEMM core: C(16x16 tile) = A[rows,K] * Bt[cols,K]^T ----------------
static __device__ __forceinline__ v8f wmma_tile_bf16(const unsigned short* __restrict__ A,
                                                     const unsigned short* __restrict__ Bt,
                                                     int rowBase, int colBase, int Kdim,
                                                     float biasv) {
  const int lane = threadIdx.x & 31;
  const int half = lane >> 4;          // 0: lanes 0-15, 1: lanes 16-31
  const int m    = lane & 15;
  const unsigned* ap = (const unsigned*)(A  + (size_t)(rowBase + m) * Kdim);
  const unsigned* bp = (const unsigned*)(Bt + (size_t)(colBase + m) * Kdim);
  v8f acc;
#pragma unroll
  for (int r = 0; r < 8; ++r) acc[r] = biasv;

  for (int kb = 0; kb < Kdim; kb += 32) {
    const int pb = kb >> 1;            // pair (dword) offset
    v8u au, bu;
    // A 16x32 bf16 frag: VGPR0-3 -> K {0..7 | 8..15}, VGPR4-7 -> K {16..23 | 24..31}
#pragma unroll
    for (int j = 0; j < 4; ++j) {
      au[j]     = ap[pb + j + 4 * half];
      au[j + 4] = ap[pb + 8 + j + 4 * half];
    }
    // B 32x16 bf16 frag: VGPR0-3 -> K {0..7 | 8..15}, VGPR4-7 -> K {16..23 | 24..31}, col = colBase + m
#pragma unroll
    for (int j = 0; j < 8; ++j) bu[j] = bp[pb + (j & 3) + 4 * half + 8 * (j >> 2)];

    v16bf a = __builtin_bit_cast(v16bf, au);
    v16bf b = __builtin_bit_cast(v16bf, bu);
    acc = __builtin_amdgcn_wmma_f32_16x16x32_bf16(false, a, false, b,
                                                  (short)0, acc, false, false);
  }
  return acc;
}

// C = A @ Bt^T + bias, fp32 out.  block=256 (8 waves), wave w -> col tile blockIdx.y*8+w
__global__ void gemm_bf16_f32_kernel(const unsigned short* __restrict__ A,
                                     const unsigned short* __restrict__ Bt,
                                     const float* __restrict__ bias,
                                     float* __restrict__ C, int Kdim, int Nout) {
  const int wave = threadIdx.x >> 5;
  const int rowBase = blockIdx.x * 16;
  const int colBase = (blockIdx.y * 8 + wave) * 16;
  const int lane = threadIdx.x & 31;
  const int half = lane >> 4;
  const int m = lane & 15;
  const int n = colBase + m;
  const float bv = bias ? bias[n] : 0.0f;
  v8f acc = wmma_tile_bf16(A, Bt, rowBase, colBase, Kdim, bv);
#pragma unroll
  for (int r = 0; r < 8; ++r)
    C[(size_t)(rowBase + r + 8 * half) * Nout + n] = acc[r];
}

// C = gelu(A @ Bt^T + bias), bf16 out (feeds next WMMA stage directly)
__global__ void gemm_bf16_gelu_bf16_kernel(const unsigned short* __restrict__ A,
                                           const unsigned short* __restrict__ Bt,
                                           const float* __restrict__ bias,
                                           unsigned short* __restrict__ C,
                                           int Kdim, int Nout) {
  const int wave = threadIdx.x >> 5;
  const int rowBase = blockIdx.x * 16;
  const int colBase = (blockIdx.y * 8 + wave) * 16;
  const int lane = threadIdx.x & 31;
  const int half = lane >> 4;
  const int m = lane & 15;
  const int n = colBase + m;
  v8f acc = wmma_tile_bf16(A, Bt, rowBase, colBase, Kdim, bias[n]);
#pragma unroll
  for (int r = 0; r < 8; ++r) {
    float x = acc[r];
    float g = 0.5f * x * (1.0f + erff(x * 0.70710678118654752f));
    C[(size_t)(rowBase + r + 8 * half) * Nout + n] = f2bf(g);
  }
}

// ---------------- edge phase ----------------
// per (edge, head): score = (q[src]·k[dst])/4 * sigmoid(MLP(edge_attr)); atomicMax for segment max
__global__ void edge_scores_kernel(const float* __restrict__ q, const float* __restrict__ k,
                                   const long long* __restrict__ ei,
                                   const float* __restrict__ ea,
                                   const float* __restrict__ We1, const float* __restrict__ be1,
                                   const float* __restrict__ We2, const float* __restrict__ be2,
                                   float* __restrict__ sc, unsigned* __restrict__ mu) {
  int idx = blockIdx.x * blockDim.x + threadIdx.x;
  if (idx >= EE * NHH) return;
  const int e = idx >> 3, h = idx & 7;
  const int src = (int)ei[e];
  const int dst = (int)ei[EE + e];
  const float4* qr = (const float4*)(q + (size_t)src * HH + h * HDD);
  const float4* kr = (const float4*)(k + (size_t)dst * HH + h * HDD);
  float dot = 0.f;
#pragma unroll
  for (int j = 0; j < 4; ++j) {
    float4 a = qr[j], b = kr[j];
    dot += a.x * b.x + a.y * b.y + a.z * b.z + a.w * b.w;
  }
  const float av = ea[e];
  float z = be2[h];
#pragma unroll
  for (int j = 0; j < 32; ++j)
    z += fmaxf(fmaf(av, We1[j], be1[j]), 0.f) * We2[j * NHH + h];
  const float ew = 1.f / (1.f + expf(-z));
  const float s = dot * 0.25f * ew;     // 1/sqrt(16)
  sc[idx] = s;
  atomicMax(&mu[(size_t)dst * NHH + h], fenc(s));
}

// e = exp(score - m[dst]); denom[dst] += e   (scores overwritten in place)
__global__ void edge_expsum_kernel(const long long* __restrict__ ei,
                                   float* __restrict__ sc,
                                   const unsigned* __restrict__ mu,
                                   float* __restrict__ den) {
  int idx = blockIdx.x * blockDim.x + threadIdx.x;
  if (idx >= EE * NHH) return;
  const int e = idx >> 3, h = idx & 7;
  const int dst = (int)ei[EE + e];
  const float ev = expf(sc[idx] - fdec(mu[(size_t)dst * NHH + h]));
  sc[idx] = ev;
  atomicAdd(&den[(size_t)dst * NHH + h], ev);
}

// agg[dst, c] += (e/denom) * v[src, c]; one thread handles 4 channels (float4)
__global__ void edge_aggregate_kernel(const long long* __restrict__ ei,
                                      const float* __restrict__ sc,
                                      const float* __restrict__ den,
                                      const float* __restrict__ v,
                                      float* __restrict__ agg) {
  int idx = blockIdx.x * blockDim.x + threadIdx.x;
  if (idx >= EE * 32) return;
  const int e = idx >> 5;
  const int c0 = (idx & 31) * 4;
  const int h = c0 >> 4;
  const int src = (int)ei[e];
  const int dst = (int)ei[EE + e];
  const float at = sc[(size_t)e * NHH + h] / den[(size_t)dst * NHH + h];
  const float4 vv = *(const float4*)(v + (size_t)src * HH + c0);
  float* ag = agg + (size_t)dst * HH + c0;
  atomicAdd(ag + 0, at * vv.x);
  atomicAdd(ag + 1, at * vv.y);
  atomicAdd(ag + 2, at * vv.z);
  atomicAdd(ag + 3, at * vv.w);
}

// ---------------- residual + layernorm (one wave32 per row) ----------------
__global__ void add_ln_kernel(const float* __restrict__ a, const float* __restrict__ b,
                              const float* __restrict__ g, const float* __restrict__ be,
                              float* __restrict__ out, unsigned short* __restrict__ outb) {
  const int row = blockIdx.x * 4 + (threadIdx.x >> 5);
  const int lane = threadIdx.x & 31;
  const float* ar = a + (size_t)row * HH;
  const float* br = b + (size_t)row * HH;
  float v[4];
  float s = 0.f;
#pragma unroll
  for (int i = 0; i < 4; ++i) {
    int c = lane + 32 * i;
    v[i] = ar[c] + br[c];
    s += v[i];
  }
  const float mean = wave_sum32(s) * (1.0f / HH);
  float q = 0.f;
#pragma unroll
  for (int i = 0; i < 4; ++i) { float d = v[i] - mean; q += d * d; }
  const float rstd = rsqrtf(wave_sum32(q) * (1.0f / HH) + 1e-5f);
#pragma unroll
  for (int i = 0; i < 4; ++i) {
    int c = lane + 32 * i;
    float o = (v[i] - mean) * rstd * g[c] + be[c];
    out[(size_t)row * HH + c] = o;
    if (outb) outb[(size_t)row * HH + c] = f2bf(o);
  }
}

// ---------------- launch ----------------
extern "C" void kernel_launch(void* const* d_in, const int* in_sizes, int n_in,
                              void* d_out, int out_size, void* d_ws, size_t ws_size,
                              hipStream_t stream) {
  (void)in_sizes; (void)n_in; (void)out_size; (void)ws_size;

  const float*     x    = (const float*)d_in[0];
  const long long* ei   = (const long long*)d_in[1];
  const float*     ea   = (const float*)d_in[2];
  const float*     Wq   = (const float*)d_in[3];
  const float*     Wk   = (const float*)d_in[4];
  const float*     Wv   = (const float*)d_in[5];
  const float*     Wo   = (const float*)d_in[6];
  const float*     bo   = (const float*)d_in[7];
  const float*     We1  = (const float*)d_in[8];
  const float*     be1  = (const float*)d_in[9];
  const float*     We2  = (const float*)d_in[10];
  const float*     be2  = (const float*)d_in[11];
  const float*     ln1g = (const float*)d_in[12];
  const float*     ln1b = (const float*)d_in[13];
  const float*     ln2g = (const float*)d_in[14];
  const float*     ln2b = (const float*)d_in[15];
  const float*     Wf1  = (const float*)d_in[16];
  const float*     bf1  = (const float*)d_in[17];
  const float*     Wf2  = (const float*)d_in[18];
  const float*     bf2  = (const float*)d_in[19];
  float* out = (float*)d_out;

  // workspace layout (bytes); all offsets 256B-aligned
  char* w = (char*)d_ws;
  const size_t SZ_XB  = (size_t)NN * HH * 2;       // x bf16; later reused as agg-bf16
  const size_t SZ_WT  = (size_t)HH * HH * 2;       // 32768
  const size_t SZ_WF  = (size_t)HH * FF * 2;       // 131072
  const size_t SZ_NH4 = (size_t)NN * HH * 4;       // 51.2 MB
  const size_t SZ_SC  = (size_t)EE * NHH * 4;      // 51.2 MB
  const size_t SZ_MH  = (size_t)NN * NHH * 4;      // 3.2 MB

  size_t off = 0;
  unsigned short* XB   = (unsigned short*)(w + off); off += SZ_XB;
  unsigned short* WQT  = (unsigned short*)(w + off); off += SZ_WT;
  unsigned short* WKT  = (unsigned short*)(w + off); off += SZ_WT;
  unsigned short* WVT  = (unsigned short*)(w + off); off += SZ_WT;
  unsigned short* WOT  = (unsigned short*)(w + off); off += SZ_WT;
  unsigned short* WF1T = (unsigned short*)(w + off); off += SZ_WF;
  unsigned short* WF2T = (unsigned short*)(w + off); off += SZ_WF;
  float*          Q    = (float*)(w + off);          off += SZ_NH4;  // reused as PROJ
  float*          K    = (float*)(w + off);          off += SZ_NH4;  // reused as FFN2OUT
  float*          V    = (float*)(w + off);          off += SZ_NH4;
  float*          SC   = (float*)(w + off);          off += SZ_SC;   // reused as X1 bf16
  unsigned*       MU   = (unsigned*)(w + off);       off += SZ_MH;
  float*          DEN  = (float*)(w + off);          off += SZ_MH;
  float*          AGG  = (float*)(w + off);          off += SZ_NH4;  // reused as X1 f32
  unsigned short* HB   = (unsigned short*)(w + off); off += (size_t)NN * FF * 2;

  unsigned short* AGGB    = XB;                   // reuse
  float*          PROJ    = Q;                    // reuse
  float*          X1      = AGG;                  // reuse (LN1 reads x & PROJ only)
  unsigned short* X1B     = (unsigned short*)SC;  // reuse
  float*          FFN2OUT = K;                    // reuse

  const int TPB = 256;
  const int nxh = NN * HH;                         // 12.8M

  // 1. conversions
  hipLaunchKernelGGL(f32_to_bf16_kernel, dim3((nxh + TPB - 1) / TPB), dim3(TPB), 0, stream, x, XB, nxh);
  hipLaunchKernelGGL(transpose_to_bf16_kernel, dim3((HH * HH + TPB - 1) / TPB), dim3(TPB), 0, stream, Wq, WQT, HH, HH);
  hipLaunchKernelGGL(transpose_to_bf16_kernel, dim3((HH * HH + TPB - 1) / TPB), dim3(TPB), 0, stream, Wk, WKT, HH, HH);
  hipLaunchKernelGGL(transpose_to_bf16_kernel, dim3((HH * HH + TPB - 1) / TPB), dim3(TPB), 0, stream, Wv, WVT, HH, HH);
  hipLaunchKernelGGL(transpose_to_bf16_kernel, dim3((HH * HH + TPB - 1) / TPB), dim3(TPB), 0, stream, Wo, WOT, HH, HH);
  hipLaunchKernelGGL(transpose_to_bf16_kernel, dim3((HH * FF + TPB - 1) / TPB), dim3(TPB), 0, stream, Wf1, WF1T, HH, FF);
  hipLaunchKernelGGL(transpose_to_bf16_kernel, dim3((FF * HH + TPB - 1) / TPB), dim3(TPB), 0, stream, Wf2, WF2T, FF, HH);

  // 2. Q/K/V projections (WMMA bf16)
  const dim3 gQKV(NN / 16, 1);
  hipLaunchKernelGGL(gemm_bf16_f32_kernel, gQKV, dim3(TPB), 0, stream, XB, WQT, (const float*)nullptr, Q, HH, HH);
  hipLaunchKernelGGL(gemm_bf16_f32_kernel, gQKV, dim3(TPB), 0, stream, XB, WKT, (const float*)nullptr, K, HH, HH);
  hipLaunchKernelGGL(gemm_bf16_f32_kernel, gQKV, dim3(TPB), 0, stream, XB, WVT, (const float*)nullptr, V, HH, HH);

  // 3. zero-init segment-max/denom/aggregation buffers
  hipMemsetAsync(MU, 0, SZ_MH, stream);
  hipMemsetAsync(DEN, 0, SZ_MH, stream);
  hipMemsetAsync(AGG, 0, SZ_NH4, stream);

  // 4. edge phase
  const int nEH = EE * NHH;                        // 12.8M
  hipLaunchKernelGGL(edge_scores_kernel, dim3(nEH / TPB), dim3(TPB), 0, stream,
                     Q, K, ei, ea, We1, be1, We2, be2, SC, MU);
  hipLaunchKernelGGL(edge_expsum_kernel, dim3(nEH / TPB), dim3(TPB), 0, stream, ei, SC, MU, DEN);
  hipLaunchKernelGGL(edge_aggregate_kernel, dim3(EE * 32 / TPB), dim3(TPB), 0, stream, ei, SC, DEN, V, AGG);

  // 5. out-projection + residual + LN1
  hipLaunchKernelGGL(f32_to_bf16_kernel, dim3((nxh + TPB - 1) / TPB), dim3(TPB), 0, stream, AGG, AGGB, nxh);
  hipLaunchKernelGGL(gemm_bf16_f32_kernel, gQKV, dim3(TPB), 0, stream, AGGB, WOT, bo, PROJ, HH, HH);
  hipLaunchKernelGGL(add_ln_kernel, dim3(NN / 4), dim3(128), 0, stream, x, PROJ, ln1g, ln1b, X1, X1B);

  // 6. FFN (WMMA bf16; GELU fused into first GEMM epilogue, bf16 intermediate)
  hipLaunchKernelGGL(gemm_bf16_gelu_bf16_kernel, dim3(NN / 16, FF / 128), dim3(TPB), 0, stream,
                     X1B, WF1T, bf1, HB, HH, FF);
  hipLaunchKernelGGL(gemm_bf16_f32_kernel, dim3(NN / 16, 1), dim3(TPB), 0, stream,
                     HB, WF2T, bf2, FFN2OUT, FF, HH);

  // 7. residual + LN2 -> output
  hipLaunchKernelGGL(add_ln_kernel, dim3(NN / 4), dim3(128), 0, stream, X1, FFN2OUT, ln2g, ln2b, out, (unsigned short*)nullptr);
}
